// SkipConnections_29703993819758
// MI455X (gfx1250) — matrix operand order repaired and checked by reading, not compiled, not run on hardware
//
#include <hip/hip_runtime.h>

#define BB 2
#define CC 64
#define NN 30000
#define VOXW 64
#define KTOT 1728          // 64 channels * 27 taps
#define KSTEPS 54          // KTOT / 32
#define LDS_STRIDE 1736    // padded row stride (ushorts) -> conflict-free ds_load_b128

typedef __attribute__((ext_vector_type(16))) __bf16 v16bf;
typedef __attribute__((ext_vector_type(8)))  float  v8f;

__device__ __forceinline__ unsigned short f2bf(float f) {
  union { float f; unsigned u; } x; x.f = f;
  unsigned u = x.u;
  unsigned r = u + 0x7FFFu + ((u >> 16) & 1u);  // round-to-nearest-even
  return (unsigned short)(r >> 16);
}

// ---------------------------------------------------------------------------
// Prep: convert conv_w (f32, [O=64][C=64][27]) into bf16 WMMA B-fragments.
// Fragment (otile, s): 32 lanes x 16 bf16; lane -> col o = otile*16+(lane&15),
// element j -> K = 32*s + (lane>>4)*16 + j, K = c*27 + tap.
// ---------------------------------------------------------------------------
__global__ void prep_w_kernel(const float* __restrict__ conv_w,
                              unsigned short* __restrict__ wsB) {
  int frag = blockIdx.x;          // otile*KSTEPS + s  (216 frags)
  int lane = threadIdx.x;         // 0..31
  int otile = frag / KSTEPS;
  int s     = frag % KSTEPS;
  int o     = otile * 16 + (lane & 15);
  int kbase = 32 * s + (lane >> 4) * 16;

  union { unsigned short h[16]; uint4 q[2]; } u;
#pragma unroll
  for (int j = 0; j < 16; ++j) {
    int K   = kbase + j;          // 0..1727
    int c   = K / 27;
    int tap = K % 27;
    u.h[j] = f2bf(conv_w[(o * CC + c) * 27 + tap]);
  }
  uint4* dst = (uint4*)(wsB + ((size_t)frag * 32 + lane) * 16);
  dst[0] = u.q[0];
  dst[1] = u.q[1];
}

// Fragment register pair used for double buffering.
struct ABfrag { union { uint4 q[2]; v16bf v; }; };

__device__ __forceinline__ void load_a_frag(const unsigned short* arow, int s,
                                            int hi, ABfrag& A) {
  A.q[0] = *(const uint4*)(arow + 32 * s + hi * 16);
  A.q[1] = *(const uint4*)(arow + 32 * s + 8 + hi * 16);
}

__device__ __forceinline__ void load_b_frag(const unsigned short* bbase, int s,
                                            ABfrag& Bm) {
  const uint4* bq = (const uint4*)(bbase + (size_t)s * (32 * 16));
  Bm.q[0] = bq[0];
  Bm.q[1] = bq[1];
}

// ---------------------------------------------------------------------------
// Fused: trilinear 27-tap sampling -> LDS (bf16 A-matrix rows) -> WMMA GEMM.
// Block = 128 threads (4 waves) owns 16 vertices; each wave owns one 16-wide
// output column tile (O = 64 = 4 tiles), K loop = 54 wmma_f32_16x16x32_bf16,
// software-pipelined one stage deep so fragment fetch overlaps the WMMA.
// ---------------------------------------------------------------------------
__global__ void __launch_bounds__(128)
fused_sample_gemm_kernel(const float* __restrict__ vox,
                         const float* __restrict__ verts,
                         const unsigned short* __restrict__ wsB,
                         const float* __restrict__ bias,
                         float* __restrict__ out) {
  __shared__ unsigned short featsLDS[16 * LDS_STRIDE];  // 55,552 bytes

  const int blk = blockIdx.x;                 // 0 .. B*(N/16)-1
  const int b   = blk / (NN / 16);
  const int n0  = (blk % (NN / 16)) * 16;
  const int t   = threadIdx.x;

  // ---------------- sampling phase: 1024 (vertex,channel) pairs ------------
#pragma unroll 1
  for (int p = 0; p < 8; ++p) {
    int pair = t + 128 * p;                   // 0..1023
    int v    = pair >> 6;                     // vertex 0..15
    int c    = pair & 63;                     // channel 0..63

    const float* vp = verts + ((size_t)b * NN + n0 + v) * 3;
    float xb = (vp[0] + 1.0f) * 31.5f;        // voxel-space base coords
    float yb = (vp[1] + 1.0f) * 31.5f;
    float zb = (vp[2] + 1.0f) * 31.5f;

    int   xi0[3], xi1[3], yi0[3], yi1[3], zi0[3], zi1[3];
    float fx[3], fy[3], fz[3];
#pragma unroll
    for (int s = 0; s < 3; ++s) {
      float sh = (float)(s - 1);
      float xc = fminf(fmaxf(xb + sh, 0.0f), 63.0f);
      float yc = fminf(fmaxf(yb + sh, 0.0f), 63.0f);
      float zc = fminf(fmaxf(zb + sh, 0.0f), 63.0f);
      float x0f = floorf(xc), y0f = floorf(yc), z0f = floorf(zc);
      xi0[s] = (int)x0f; yi0[s] = (int)y0f; zi0[s] = (int)z0f;
      xi1[s] = xi0[s] + 1 > 63 ? 63 : xi0[s] + 1;
      yi1[s] = yi0[s] + 1 > 63 ? 63 : yi0[s] + 1;
      zi1[s] = zi0[s] + 1 > 63 ? 63 : zi0[s] + 1;
      fx[s] = xc - x0f; fy[s] = yc - y0f; fz[s] = zc - z0f;
    }

    const float* volc = vox + ((size_t)(b * CC + c)) * (VOXW * VOXW * VOXW);
    unsigned short* row = &featsLDS[v * LDS_STRIDE + c * 27];

#pragma unroll
    for (int tap = 0; tap < 27; ++tap) {
      const int sx = tap / 9, sy = (tap / 3) % 3, sz = tap % 3;
      const int x0 = xi0[sx], x1 = xi1[sx];
      const float wx1 = fx[sx], wx0 = 1.0f - wx1;
      const float wy1 = fy[sy], wy0 = 1.0f - wy1;
      const float wz1 = fz[sz], wz0 = 1.0f - wz1;

      const float* b00 = volc + (zi0[sz] * 64 + yi0[sy]) * 64;
      const float* b01 = volc + (zi0[sz] * 64 + yi1[sy]) * 64;
      const float* b10 = volc + (zi1[sz] * 64 + yi0[sy]) * 64;
      const float* b11 = volc + (zi1[sz] * 64 + yi1[sy]) * 64;

      float r = ((b00[x0] * wx0 + b00[x1] * wx1) * wy0 +
                 (b01[x0] * wx0 + b01[x1] * wx1) * wy1) * wz0 +
                ((b10[x0] * wx0 + b10[x1] * wx1) * wy0 +
                 (b11[x0] * wx0 + b11[x1] * wx1) * wy1) * wz1;

      row[tap] = f2bf(r);
    }
  }

  __syncthreads();

  // ---------------- GEMM phase: 4 waves x (16 vtx x 16 out) tiles ----------
  const int lane = t & 31;
  const int wv   = t >> 5;            // wave id == output column tile 0..3
  const int r    = lane & 15;         // A row / D column index
  const int hi   = lane >> 4;

  // Hoist bias load so the tail isn't a dependent global-load stall.
  const float bv = bias[wv * 16 + r];

  const unsigned short* arow  = &featsLDS[r * LDS_STRIDE];
  const unsigned short* bbase = wsB + ((size_t)(wv * KSTEPS) * 32 + lane) * 16;

  v8f acc = {};
  ABfrag A0, B0, A1, B1;

  // Software pipeline: fetch frag s+1 while WMMA consumes frag s.
  load_a_frag(arow, 0, hi, A0);
  load_b_frag(bbase, 0, B0);

#pragma unroll 1
  for (int s = 0; s + 2 < KSTEPS; s += 2) {
    load_a_frag(arow, s + 1, hi, A1);
    load_b_frag(bbase, s + 1, B1);
    acc = __builtin_amdgcn_wmma_f32_16x16x32_bf16(
        false, A0.v, false, B0.v, (short)0, acc, false, false);

    load_a_frag(arow, s + 2, hi, A0);
    load_b_frag(bbase, s + 2, B0);
    acc = __builtin_amdgcn_wmma_f32_16x16x32_bf16(
        false, A1.v, false, B1.v, (short)0, acc, false, false);
  }
  // Epilogue: steps KSTEPS-2 (in A0/B0) and KSTEPS-1.
  load_a_frag(arow, KSTEPS - 1, hi, A1);
  load_b_frag(bbase, KSTEPS - 1, B1);
  acc = __builtin_amdgcn_wmma_f32_16x16x32_bf16(
      false, A0.v, false, B0.v, (short)0, acc, false, false);
  acc = __builtin_amdgcn_wmma_f32_16x16x32_bf16(
      false, A1.v, false, B1.v, (short)0, acc, false, false);

  // bias + store: D element i -> vertex (i + 8*hi), output col wv*16 + r
  float* outp = out + ((size_t)b * NN + n0) * 64 + wv * 16 + r;
#pragma unroll
  for (int i = 0; i < 8; ++i) {
    int vtx = i + 8 * hi;
    outp[(size_t)vtx * 64] = acc[i] + bv;
  }
}

extern "C" void kernel_launch(void* const* d_in, const int* in_sizes, int n_in,
                              void* d_out, int out_size, void* d_ws, size_t ws_size,
                              hipStream_t stream) {
  const float* vox    = (const float*)d_in[0];  // (B, C, D, H, W) f32
  const float* verts  = (const float*)d_in[1];  // (B, N, 3) f32
  const float* conv_w = (const float*)d_in[2];  // (C, C, 27) f32
  const float* conv_b = (const float*)d_in[3];  // (C,) f32
  unsigned short* wsB = (unsigned short*)d_ws;  // 4*54*32*16 bf16 = 221,184 B

  prep_w_kernel<<<4 * KSTEPS, 32, 0, stream>>>(conv_w, wsB);
  fused_sample_gemm_kernel<<<BB * (NN / 16), 128, 0, stream>>>(
      vox, verts, wsB, conv_b, (float*)d_out);
}
